// GTM_SM_45183055954123
// MI455X (gfx1250) — compile-verified, matrix operand order checked by reading, not run on hardware
//
#include <hip/hip_runtime.h>

// ---------------- problem constants ----------------
constexpr int   kB  = 2048;
constexpr int   kTO = 256;
constexpr int   kTP = 32;
constexpr int   kT  = kTO + kTP;        // 288
constexpr int   kA  = 5;
constexpr int   kZ  = 16;
constexpr int   kK  = 5;
constexpr float kDelta = 1e-4f;
constexpr float kRstd  = 1e-3f;

// ---------------- workspace layout (bytes) ----------------
constexpr size_t OFF_STOBS  = 0;                                          // (TO,B,2) f32
constexpr size_t OFF_STPRED = OFF_STOBS  + (size_t)kTO * kB * 2 * 4;      // (TP,B,2) f32
constexpr size_t OFF_ZM     = OFF_STPRED + (size_t)kTP * kB * 2 * 4;      // (B,TO,16) f32
constexpr size_t OFF_ZS     = OFF_ZM     + (size_t)kB * kTO * 16 * 4;     // (B,TO,16) f32
constexpr size_t OFF_Z      = OFF_ZS     + (size_t)kB * kTO * 16 * 4;     // (B,TP,16) f32
// fragment-ready f16 weights: [(ct*KC + kc)*32 + lane]*16 + i  (32B per lane-frag)
constexpr size_t OFF_ENCWF  = OFF_Z      + (size_t)kB * kTP * 16 * 4;     // 6*32*512 h
constexpr size_t OFF_MEANWF = OFF_ENCWF  + (size_t)6 * 32 * 512 * 2;      // 16*1*512 h
constexpr size_t OFF_STDWF  = OFF_MEANWF + (size_t)16 * 512 * 2;          // 16*1*512 h
constexpr size_t OFF_DW1F   = OFF_STDWF  + (size_t)16 * 512 * 2;          // 1*32*512 h (K padded)
constexpr size_t OFF_DW2F   = OFF_DW1F   + (size_t)32 * 512 * 2;          // 16*12*512 h

// ---------------- vector types & WMMA helpers ----------------
typedef __attribute__((ext_vector_type(16))) _Float16 v16h;
typedef __attribute__((ext_vector_type(8)))  float    v8f;

__device__ __forceinline__ v8f wmma_f16(v16h a, v16h b, v8f c) {
  // D = A(16x32 f16) * B(32x16 f16) + C(16x16 f32)
  return __builtin_amdgcn_wmma_f32_16x16x32_f16(false, a, false, b, (short)0, c,
                                                false, false);
}

__device__ __forceinline__ v8f vzero8() {
  v8f z;
#pragma unroll
  for (int i = 0; i < 8; ++i) z[i] = 0.f;
  return z;
}

// WMMA lane-order K offset for fragment element i of a lane-half hh:
// i<8 -> 8*hh+i  (K 0..15) ; i>=8 -> 16+8*hh+(i-8) (K 16..31)
__device__ __forceinline__ int koff_of(int hh, int i) {
  return (i < 8) ? (8 * hh + i) : (16 + 8 * hh + (i - 8));
}

// inverse: matrix K-offset rem (0..31) -> (hh, i)
__device__ __forceinline__ void inv_koff(int rem, int& hh, int& i) {
  int seg = rem >> 3, sub = rem & 7;
  hh = seg & 1;
  i  = sub + ((seg >> 1) << 3);
}

// load one fragment (32 contiguous bytes per lane) from fragment-ready storage
__device__ __forceinline__ v16h fragLoad(const _Float16* F, int tile, int kc,
                                         int KC, int lane) {
  return *(const v16h*)(F + (((size_t)(tile * KC + kc) * 32 + lane) << 4));
}

// ---------------- 0) f32 -> f16 fragment-ready weight shuffle ----------------
__device__ __forceinline__ void shuffle_one(const float* __restrict__ W,
                                            _Float16* __restrict__ F, int didx,
                                            int KC, int N, int Kreal) {
  int i    = didx & 15;
  int lane = (didx >> 4) & 31;
  int rest = didx >> 9;
  int kc   = rest % KC;
  int ct   = rest / KC;
  int hh   = lane >> 4;
  int col  = ct * 16 + (lane & 15);
  int k    = kc * 32 + koff_of(hh, i);
  F[didx]  = (k < Kreal) ? (_Float16)W[(size_t)k * N + col] : (_Float16)0.f;
}

__global__ void prep_kernel(const float* __restrict__ enc_w,
                            const float* __restrict__ mean_w,
                            const float* __restrict__ std_w,
                            const float* __restrict__ dec_w1,
                            const float* __restrict__ dec_w2,
                            _Float16* __restrict__ encw_f,
                            _Float16* __restrict__ meanw_f,
                            _Float16* __restrict__ stdw_f,
                            _Float16* __restrict__ dw1f,
                            _Float16* __restrict__ dw2f) {
  int i = blockIdx.x * blockDim.x + threadIdx.x;
  if (i < 6 * 32 * 512) {                       // enc_w (192,512): KC=6, NT=32
    shuffle_one(enc_w, encw_f, i, 6, 512, 192);
  }
  if (i < 16 * 12 * 512) {                      // dec_w2 (512,192): KC=16, NT=12
    shuffle_one(dec_w2, dw2f, i, 16, 192, 512);
  }
  if (i < 1 * 32 * 512) {                       // dec_w1 (16,512): K padded to 32
    shuffle_one(dec_w1, dw1f, i, 1, 512, 16);
  }
  if (i < 16 * 1 * 512) {                       // mean_w / std_w (512,16)
    shuffle_one(mean_w, meanw_f, i, 16, 16, 512);
    shuffle_one(std_w,  stdw_f,  i, 16, 16, 512);
  }
}

// ---------------- 1) sequential state recurrence ----------------
__global__ void recur_kernel(const int* __restrict__ actions,
                             const float* __restrict__ noise_obs,
                             const float* __restrict__ noise_pred,
                             const float* __restrict__ A_w,
                             const float* __restrict__ sw1, const float* __restrict__ sb1,
                             const float* __restrict__ sw2, const float* __restrict__ sb2,
                             float* __restrict__ st_obs, float* __restrict__ st_pred) {
  int b = blockIdx.x * blockDim.x + threadIdx.x;
  if (b >= kB) return;
  float aw0[kA], aw1[kA], W1a[10], W1b[10], B1[10], W2a[10], W2b[10], B2a, B2b;
#pragma unroll
  for (int i = 0; i < kA; ++i) { aw0[i] = A_w[i * 2]; aw1[i] = A_w[i * 2 + 1]; }
#pragma unroll
  for (int j = 0; j < 10; ++j) {
    W1a[j] = sw1[j]; W1b[j] = sw1[10 + j]; B1[j] = sb1[j];
    W2a[j] = sw2[j * 2]; W2b[j] = sw2[j * 2 + 1];
  }
  B2a = sb2[0]; B2b = sb2[1];

  float s0 = 0.f, s1 = 0.f;
  st_obs[b * 2 + 0] = 0.f;  // t = 0
  st_obs[b * 2 + 1] = 0.f;
  for (int t = 1; t < kTO; ++t) {
    int a = actions[b * kT + t];
    float r0 = aw0[a], r1 = aw1[a];
    float u0 = s0 + r0, u1 = s1 + r1;
    float m0 = B2a, m1 = B2b;
#pragma unroll
    for (int j = 0; j < 10; ++j) {
      float v = tanhf(u0 * W1a[j] + u1 * W1b[j] + B1[j]);
      m0 += v * W2a[j]; m1 += v * W2b[j];
    }
    m0 = 1.f / (1.f + expf(-m0));
    m1 = 1.f / (1.f + expf(-m1));
    s0 += r0 * m0 + noise_obs[((size_t)t * kB + b) * 2 + 0] * kRstd;
    s1 += r1 * m1 + noise_obs[((size_t)t * kB + b) * 2 + 1] * kRstd;
    st_obs[((size_t)t * kB + b) * 2 + 0] = s0;
    st_obs[((size_t)t * kB + b) * 2 + 1] = s1;
  }
  for (int tp = 0; tp < kTP; ++tp) {
    int a = actions[b * kT + kTO + tp];
    float r0 = aw0[a], r1 = aw1[a];
    float u0 = s0 + r0, u1 = s1 + r1;
    float m0 = B2a, m1 = B2b;
#pragma unroll
    for (int j = 0; j < 10; ++j) {
      float v = tanhf(u0 * W1a[j] + u1 * W1b[j] + B1[j]);
      m0 += v * W2a[j]; m1 += v * W2b[j];
    }
    m0 = 1.f / (1.f + expf(-m0));
    m1 = 1.f / (1.f + expf(-m1));
    s0 += r0 * m0 + noise_pred[((size_t)tp * kB + b) * 2 + 0] * kRstd;
    s1 += r1 * m1 + noise_pred[((size_t)tp * kB + b) * 2 + 1] * kRstd;
    st_pred[((size_t)tp * kB + b) * 2 + 0] = s0;
    st_pred[((size_t)tp * kB + b) * 2 + 1] = s1;
  }
}

// ---------------- 2) fused patch encoder (WMMA) ----------------
// block = 256 threads (8 waves), owns 16 rows (row = b*TO + to).
__global__ __launch_bounds__(256) void enc_kernel(
    const float* __restrict__ x, const int* __restrict__ position,
    const _Float16* __restrict__ encw_f, const float* __restrict__ enc_b,
    const _Float16* __restrict__ meanw_f, const float* __restrict__ mean_b,
    const _Float16* __restrict__ stdw_f, const float* __restrict__ std_b,
    float* __restrict__ zm, float* __restrict__ zs) {
  __shared__ __align__(32) _Float16 Af[6 * 32 * 16];    // patches, fragment-ready
  __shared__ __align__(32) _Float16 Hf[16 * 32 * 16];   // h tile, fragment-ready
  __shared__ int ph[16], pw[16], pbv[16];

  const int tid = threadIdx.x;
  const int lane = tid & 31;
  const int wave = tid >> 5;
  const int r0 = blockIdx.x * 16;

  if (tid < 16) {
    int r = r0 + tid;
    int b = r / kTO, to = r % kTO;
    pbv[tid] = b;
    ph[tid] = position[(size_t)b * 2 * kT + to];        // position[b,0,to]
    pw[tid] = position[(size_t)b * 2 * kT + kT + to];   // position[b,1,to]
  }
  __syncthreads();

  // stage patches directly in fragment order (dst-indexed)
  for (int e = tid; e < 6 * 32 * 16; e += 256) {
    int i  = e & 15;
    int ln = (e >> 4) & 31;
    int kc = e >> 9;
    int hh = ln >> 4, lr = ln & 15;
    int f  = kc * 32 + koff_of(hh, i);          // flat patch index 0..191
    int c = f >> 6, i8 = (f >> 3) & 7, j8 = f & 7;
    int b = pbv[lr];
    float v = x[(((size_t)b * 3 + c) * 32 + 3 * ph[lr] + i8) * 32 + 3 * pw[lr] + j8];
    Af[e] = (_Float16)(2.f * v - 1.f);
  }
  __syncthreads();

  if (lane == 0) __builtin_prefetch(encw_f, 0, 3);

  v8f acc[4];
#pragma unroll
  for (int t = 0; t < 4; ++t) acc[t] = vzero8();

#pragma unroll
  for (int kc = 0; kc < 6; ++kc) {
    v16h a = *(const v16h*)&Af[((kc * 32) + lane) << 4];
#pragma unroll
    for (int t = 0; t < 4; ++t) {
      int ct = wave * 4 + t;
      v16h b = fragLoad(encw_f, ct, kc, 6, lane);
      acc[t] = wmma_f16(a, b, acc[t]);
    }
  }

  // bias + tanh -> Hf in fragment order (K of next GEMM = column index)
  {
    const int hh_w = lane >> 4, col15 = lane & 15;
#pragma unroll
    for (int t = 0; t < 4; ++t) {
      int col = (wave * 4 + t) * 16 + col15;
      int kc = col >> 5, rem = col & 31;
      int hh_s, i_s;
      inv_koff(rem, hh_s, i_s);
      float bias = enc_b[col];
#pragma unroll
      for (int r = 0; r < 8; ++r) {
        int M = r + 8 * hh_w;                   // row of h tile
        Hf[(((kc * 32) + (hh_s << 4) + M) << 4) + i_s] =
            (_Float16)tanhf(acc[t][r] + bias);
      }
    }
  }
  __syncthreads();

  // zm / zs heads: wave 0 -> mean, wave 1 -> std
  if (wave < 2) {
    const _Float16* Wf = (wave == 0) ? meanw_f : stdw_f;
    v8f zacc = vzero8();
#pragma unroll
    for (int kc = 0; kc < 16; ++kc) {
      v16h a = *(const v16h*)&Hf[((kc * 32) + lane) << 4];
      v16h b = fragLoad(Wf, 0, kc, 16, lane);
      zacc = wmma_f16(a, b, zacc);
    }
    const int hh = lane >> 4, col = lane & 15;
    float bias = (wave == 0 ? mean_b : std_b)[col];
#pragma unroll
    for (int r = 0; r < 8; ++r) {
      int grow = r0 + r + 8 * hh;               // row = b*TO + to
      float v = zacc[r] + bias;
      if (wave == 0) zm[(size_t)grow * 16 + col] = v;
      else           zs[(size_t)grow * 16 + col] = expf(v);
    }
  }
}

// ---------------- 3) per-timestep KNN + latent sample ----------------
__global__ void knn_kernel(const float* __restrict__ st_obs,
                           const float* __restrict__ st_pred,
                           const float* __restrict__ zm, const float* __restrict__ zs,
                           const float* __restrict__ eps_z,
                           float* __restrict__ zout) {
  int g = blockIdx.x * blockDim.x + threadIdx.x;
  if (g >= kB * kTP) return;
  int tp = g / kB, b = g % kB;   // adjacent threads -> adjacent b (coalesced)

  float sp0 = st_pred[((size_t)tp * kB + b) * 2 + 0];
  float sp1 = st_pred[((size_t)tp * kB + b) * 2 + 1];

  float val[kK];
  int   idx[kK];
#pragma unroll
  for (int k = 0; k < kK; ++k) { val[k] = 3.4e38f; idx[k] = 0; }

  for (int to = 0; to < kTO; ++to) {
    float d0 = sp0 - st_obs[((size_t)to * kB + b) * 2 + 0];
    float d1 = sp1 - st_obs[((size_t)to * kB + b) * 2 + 1];
    float d = d0 * d0 + d1 * d1;
    if (d < val[kK - 1]) {
      val[kK - 1] = d; idx[kK - 1] = to;
#pragma unroll
      for (int k = kK - 1; k > 0; --k) {
        if (val[k] < val[k - 1]) {
          float tv = val[k]; val[k] = val[k - 1]; val[k - 1] = tv;
          int   ti = idx[k]; idx[k] = idx[k - 1]; idx[k - 1] = ti;
        }
      }
    }
  }

  float w[kK], tot = 0.f;
#pragma unroll
  for (int k = 0; k < kK; ++k) { w[k] = 1.f / (val[k] + kDelta); tot += w[k]; }
  float inv = 1.f / tot;
#pragma unroll
  for (int k = 0; k < kK; ++k) w[k] *= inv;

  for (int j = 0; j < kZ; ++j) {
    float zmp = 0.f, zsp = 0.f;
#pragma unroll
    for (int k = 0; k < kK; ++k) {
      size_t base = ((size_t)b * kTO + idx[k]) * 16 + j;
      zmp += w[k] * zm[base];
      zsp += w[k] * zs[base];
    }
    float e = eps_z[((size_t)tp * kB + b) * 16 + j];
    zout[((size_t)b * kTP + tp) * 16 + j] = zmp + zsp * e;
  }
}

// ---------------- 4) fused decoder (WMMA) ----------------
// block owns 16 rows (row = b*TP + tp).
__global__ __launch_bounds__(256) void dec_kernel(
    const float* __restrict__ z,
    const _Float16* __restrict__ w1f, const float* __restrict__ b1,
    const _Float16* __restrict__ w2f, const float* __restrict__ b2,
    float* __restrict__ out) {
  __shared__ __align__(32) _Float16 Zf[32 * 16];        // z tile frag (K 16->32 pad)
  __shared__ __align__(32) _Float16 Hdf[16 * 32 * 16];  // hd tile, fragment-ready

  const int tid = threadIdx.x;
  const int lane = tid & 31;
  const int wave = tid >> 5;
  const int r0 = blockIdx.x * 16;

  // stage z in fragment order (dst-indexed); K >= 16 is zero padding
  for (int e = tid; e < 32 * 16; e += 256) {
    int i  = e & 15;
    int ln = e >> 4;
    int hh = ln >> 4, lr = ln & 15;
    int k  = koff_of(hh, i);
    Zf[e] = (k < 16) ? (_Float16)z[(size_t)(r0 + lr) * 16 + k] : (_Float16)0.f;
  }
  __syncthreads();

  if (lane == 0) __builtin_prefetch(w2f, 0, 3);

  // hd = tanh(z @ dec_w1 + b1) : single K=32 WMMA per column tile
  {
    v16h a = *(const v16h*)&Zf[lane << 4];
    const int hh_w = lane >> 4, col15 = lane & 15;
#pragma unroll
    for (int t = 0; t < 4; ++t) {
      int ct = wave * 4 + t;
      v16h b = fragLoad(w1f, ct, 0, 1, lane);
      v8f c = wmma_f16(a, b, vzero8());
      int col = ct * 16 + col15;
      int kc = col >> 5, rem = col & 31;
      int hh_s, i_s;
      inv_koff(rem, hh_s, i_s);
      float bias = b1[col];
#pragma unroll
      for (int r = 0; r < 8; ++r) {
        int M = r + 8 * hh_w;
        Hdf[(((kc * 32) + (hh_s << 4) + M) << 4) + i_s] =
            (_Float16)tanhf(c[r] + bias);
      }
    }
  }
  __syncthreads();

  v8f acc[2];
  acc[0] = vzero8(); acc[1] = vzero8();

#pragma unroll
  for (int kc = 0; kc < 16; ++kc) {
    v16h a = *(const v16h*)&Hdf[((kc * 32) + lane) << 4];
#pragma unroll
    for (int t = 0; t < 2; ++t) {
      int ct = wave + 8 * t;          // wave-uniform -> EXEC stays all-ones
      if (ct < 12) {
        v16h b = fragLoad(w2f, ct, kc, 16, lane);
        acc[t] = wmma_f16(a, b, acc[t]);
      }
    }
  }

  // write (o+1)/2, transposing row = b*TP+tp -> out[(tp*B + b)*192 + col]
  {
    const int hh = lane >> 4, col15 = lane & 15;
#pragma unroll
    for (int t = 0; t < 2; ++t) {
      int ct = wave + 8 * t;
      if (ct < 12) {
        int col = ct * 16 + col15;
        float bias = b2[col];
#pragma unroll
        for (int r = 0; r < 8; ++r) {
          int rg = r0 + r + 8 * hh;
          int bb = rg >> 5;           // rg / TP
          int tp = rg & 31;           // rg % TP
          float o = tanhf(acc[t][r] + bias);
          out[((size_t)tp * kB + bb) * 192 + col] = (o + 1.f) * 0.5f;
        }
      }
    }
  }
}

// ---------------- launcher ----------------
extern "C" void kernel_launch(void* const* d_in, const int* in_sizes, int n_in,
                              void* d_out, int out_size, void* d_ws, size_t ws_size,
                              hipStream_t stream) {
  (void)in_sizes; (void)n_in; (void)out_size; (void)ws_size;
  const float* x          = (const float*)d_in[0];
  const int*   actions    = (const int*)d_in[1];
  const int*   position   = (const int*)d_in[2];
  const float* noise_obs  = (const float*)d_in[3];
  const float* noise_pred = (const float*)d_in[4];
  const float* eps_z      = (const float*)d_in[5];
  const float* enc_w      = (const float*)d_in[6];
  const float* enc_b      = (const float*)d_in[7];
  const float* mean_w     = (const float*)d_in[8];
  const float* mean_b     = (const float*)d_in[9];
  const float* std_w      = (const float*)d_in[10];
  const float* std_b      = (const float*)d_in[11];
  const float* A_w        = (const float*)d_in[12];
  const float* sw1        = (const float*)d_in[13];
  const float* sb1        = (const float*)d_in[14];
  const float* sw2        = (const float*)d_in[15];
  const float* sb2        = (const float*)d_in[16];
  const float* dec_w1     = (const float*)d_in[17];
  const float* dec_b1     = (const float*)d_in[18];
  const float* dec_w2     = (const float*)d_in[19];
  const float* dec_b2     = (const float*)d_in[20];
  float* out = (float*)d_out;

  char* ws = (char*)d_ws;
  float*    st_obs  = (float*)(ws + OFF_STOBS);
  float*    st_pred = (float*)(ws + OFF_STPRED);
  float*    zm      = (float*)(ws + OFF_ZM);
  float*    zs      = (float*)(ws + OFF_ZS);
  float*    zlat    = (float*)(ws + OFF_Z);
  _Float16* encw_f  = (_Float16*)(ws + OFF_ENCWF);
  _Float16* meanw_f = (_Float16*)(ws + OFF_MEANWF);
  _Float16* stdw_f  = (_Float16*)(ws + OFF_STDWF);
  _Float16* dw1f    = (_Float16*)(ws + OFF_DW1F);
  _Float16* dw2f    = (_Float16*)(ws + OFF_DW2F);

  prep_kernel<<<(6 * 32 * 512 + 255) / 256, 256, 0, stream>>>(
      enc_w, mean_w, std_w, dec_w1, dec_w2, encw_f, meanw_f, stdw_f, dw1f, dw2f);

  recur_kernel<<<(kB + 255) / 256, 256, 0, stream>>>(
      actions, noise_obs, noise_pred, A_w, sw1, sb1, sw2, sb2, st_obs, st_pred);

  enc_kernel<<<(kB * kTO) / 16, 256, 0, stream>>>(
      x, position, encw_f, enc_b, meanw_f, mean_b, stdw_f, std_b, zm, zs);

  knn_kernel<<<(kB * kTP + 255) / 256, 256, 0, stream>>>(
      st_obs, st_pred, zm, zs, eps_z, zlat);

  dec_kernel<<<(kB * kTP) / 16, 256, 0, stream>>>(
      zlat, dw1f, dec_b1, dw2f, dec_b2, out);
}
